// MultiCharacterConditioner_60790967107677
// MI455X (gfx1250) — matrix operand order repaired and checked by reading, not compiled, not run on hardware
//
#include <hip/hip_runtime.h>
#include <hip/hip_bf16.h>

typedef __bf16 bf16;
typedef __attribute__((ext_vector_type(16))) __bf16 v16bf;
typedef __attribute__((ext_vector_type(8)))  __bf16 v8bf;
typedef __attribute__((ext_vector_type(4)))  __bf16 v4bf;
typedef __attribute__((ext_vector_type(8)))  float  v8f;

#define WMMA_BF16(a,b,c) \
  __builtin_amdgcn_wmma_f32_16x16x32_bf16(false,(a),false,(b),(short)0,(c),false,false)

constexpr int B_ = 2, N_ = 2048, D_ = 1024, H_ = 16, HD_ = 64, C_ = 4;
constexpr float EPS_ = 1e-6f, BOOST_ = 0.3f;

// ---- workspace layout (bytes) ----
constexpr size_t QK_BYTES = (size_t)B_ * H_ * N_ * HD_ * sizeof(bf16); // 8 MiB
constexpr size_t OFF_Q  = 0;
constexpr size_t OFF_K  = OFF_Q + QK_BYTES;
constexpr size_t OFF_V  = OFF_K + QK_BYTES;                    // V transposed [b][h][hd][n]
constexpr size_t OFF_AO = OFF_V + QK_BYTES;                    // attn out bf16 [b][n][d]
constexpr size_t OFF_RM = OFF_AO + (size_t)B_ * N_ * D_ * sizeof(bf16);

// A-fragment (16x32, M=row per lane): elements 0..7 -> K=kb0A+e, 8..15 -> K=kb0A+16+e
__device__ inline v16bf load_afrag(const bf16* rowbase, int kb0A) {
  v8bf lo = *(const v8bf*)(rowbase + kb0A);
  v8bf hi = *(const v8bf*)(rowbase + kb0A + 16);
  v16bf r;
#pragma unroll
  for (int e = 0; e < 8; ++e) { r[e] = lo[e]; r[8 + e] = hi[e]; }
  return r;
}
// B-fragment (32x16, N=col per lane): elements 0..7 -> K=kbB+e, 8..15 -> K=kbB+16+e
__device__ inline v16bf load_bfrag(const bf16* rowbase, int kbB) {
  v8bf lo = *(const v8bf*)(rowbase + kbB);
  v8bf hi = *(const v8bf*)(rowbase + kbB + 16);
  v16bf r;
#pragma unroll
  for (int e = 0; e < 8; ++e) { r[e] = lo[e]; r[8 + e] = hi[e]; }
  return r;
}
__device__ inline v8f vzero8() {
  v8f z = {0.f, 0.f, 0.f, 0.f, 0.f, 0.f, 0.f, 0.f};
  return z;
}
__device__ inline v4bf cvt4(float4 f) {
  v4bf p;
  p[0] = (bf16)f.x; p[1] = (bf16)f.y; p[2] = (bf16)f.z; p[3] = (bf16)f.w;
  return p;
}

// ============ Kernel 1: QKV projection + RMSNorm(Q,K) + head split ============
// grid (B*N/16, H), block 128 (4 waves). Wave w owns head-dim cols [16w,16w+16).
__global__ __launch_bounds__(128) void qkv_kernel(
    const float* __restrict__ x, const float* __restrict__ Wqkv,
    const float* __restrict__ qw, const float* __restrict__ kw,
    bf16* __restrict__ q_ws, bf16* __restrict__ k_ws, bf16* __restrict__ v_wsT)
{
  __shared__ __align__(64) bf16 lA[16][32];     // x tile (bf16)
  __shared__ __align__(64) bf16 lB[192][32];    // W rows for q/k/v (3 * 64 cols)
  __shared__ float lO[3][16][64];               // fp32 output tiles
  __shared__ float lS[2][16];                   // rms scales

  const int tid = threadIdx.x, wave = tid >> 5, lane = tid & 31;
  const int nt = blockIdx.x % (N_ / 16), b = blockIdx.x / (N_ / 16);
  const int h = blockIdx.y;
  const int n0 = nt * 16;
  const int ar = lane & 15;
  const int kb0A = (lane < 16) ? 0 : 8;
  const int kbB  = (lane < 16) ? 0 : 8;

  // staging indices (vectorized: one float4 -> one 8-byte bf16 LDS store)
  const int sr  = tid >> 3;              // 0..15 (lA row)
  const int sc4 = (tid & 7) * 4;         // 0..28 (lA col group)
  const float* xrow = x + ((size_t)(b * N_ + n0 + sr)) * D_ + sc4;

  // per-thread W staging addresses are loop-invariant except for kk
  int wr[12], wc4[12];
#pragma unroll
  for (int j = 0; j < 12; ++j) {
    const int i = tid + j * 128;
    wr[j] = i >> 3; wc4[j] = (i & 7) * 4;
  }

  v8f acc[3];
#pragma unroll
  for (int t = 0; t < 3; ++t) acc[t] = vzero8();

  for (int kk = 0; kk < D_ / 32; ++kk) {
    // --- phase 1: issue all staging loads back-to-back (no intervening waits)
    float4 fx = *(const float4*)(xrow + kk * 32);
    float4 fw[12];
#pragma unroll
    for (int j = 0; j < 12; ++j) {
      const int r = wr[j];
      const int t = r >> 6, ch = r & 63;
      fw[j] = *(const float4*)(Wqkv + ((size_t)(t * D_ + h * HD_ + ch)) * D_ +
                               kk * 32 + wc4[j]);
    }
    // --- phase 2: convert + LDS stores (waits amortized across the batch)
    *(v4bf*)&lA[sr][sc4] = cvt4(fx);
#pragma unroll
    for (int j = 0; j < 12; ++j)
      *(v4bf*)&lB[wr[j]][wc4[j]] = cvt4(fw[j]);
    __syncthreads();

    v16bf a;
#pragma unroll
    for (int e = 0; e < 8; ++e) { a[e] = lA[ar][kb0A + e]; a[8 + e] = lA[ar][kb0A + 16 + e]; }

#pragma unroll
    for (int t = 0; t < 3; ++t) {
      const int rowi = t * 64 + wave * 16 + ar;
      v16bf bt;
#pragma unroll
      for (int i2 = 0; i2 < 16; ++i2) bt[i2] = lB[rowi][kbB + (i2 < 8 ? i2 : 8 + i2)];
      acc[t] = WMMA_BF16(a, bt, acc[t]);
    }
    __syncthreads();
  }

  const int hi8 = (lane >= 16) ? 8 : 0;
#pragma unroll
  for (int t = 0; t < 3; ++t)
#pragma unroll
    for (int r = 0; r < 8; ++r)
      lO[t][r + hi8][wave * 16 + ar] = acc[t][r];
  __syncthreads();

  if (tid < 16) {
    float sq = 0.f, sk = 0.f;
    for (int c = 0; c < HD_; ++c) {
      float qv = lO[0][tid][c], kv = lO[1][tid][c];
      sq += qv * qv; sk += kv * kv;
    }
    lS[0][tid] = rsqrtf(sq / HD_ + EPS_);
    lS[1][tid] = rsqrtf(sk / HD_ + EPS_);
  }
  __syncthreads();

  const int bh = b * H_ + h;
  // q,k writeback: 128 threads * 8 contiguous cols = 16x64, 16B stores
  {
    const int r = tid >> 3, c8 = (tid & 7) * 8;
    const float sqv = lS[0][r], skv = lS[1][r];
    v8bf pq, pk;
#pragma unroll
    for (int e = 0; e < 8; ++e) {
      pq[e] = (bf16)(lO[0][r][c8 + e] * sqv * qw[c8 + e]);
      pk[e] = (bf16)(lO[1][r][c8 + e] * skv * kw[c8 + e]);
    }
    const size_t qi = ((size_t)(bh * N_ + n0 + r)) * HD_ + c8;
    *(v8bf*)(q_ws + qi) = pq;
    *(v8bf*)(k_ws + qi) = pk;
  }
  // v transposed writeback (stride-N scatter, scalar b16)
  for (int i = tid; i < 16 * 64; i += 128) {
    const int r = i >> 6, c = i & 63;
    v_wsT[((size_t)(bh * HD_ + c)) * N_ + (n0 + r)] = (bf16)lO[2][r][c];
  }
}

// ============ Kernel 2: row max of same-character matrix ============
__global__ __launch_bounds__(256) void rowmax_kernel(const float* __restrict__ cm,
                                                     float* __restrict__ rm)
{
  int idx = blockIdx.x * blockDim.x + threadIdx.x;   // 0..B*N-1
  if (idx >= B_ * N_) return;
  int b = idx / N_, q = idx % N_;
  float c0 = cm[(b * C_ + 0) * N_ + q];
  float c1 = cm[(b * C_ + 1) * N_ + q];
  float c2 = cm[(b * C_ + 2) * N_ + q];
  float c3 = cm[(b * C_ + 3) * N_ + q];
  const float4* r0 = (const float4*)(cm + (b * C_ + 0) * N_);
  const float4* r1 = (const float4*)(cm + (b * C_ + 1) * N_);
  const float4* r2 = (const float4*)(cm + (b * C_ + 2) * N_);
  const float4* r3 = (const float4*)(cm + (b * C_ + 3) * N_);
  float mx = 0.f;
  for (int k4 = 0; k4 < N_ / 4; ++k4) {
    float4 a = r0[k4], bb = r1[k4], c = r2[k4], d = r3[k4];
    float s0 = c0 * a.x + c1 * bb.x + c2 * c.x + c3 * d.x;
    float s1 = c0 * a.y + c1 * bb.y + c2 * c.y + c3 * d.y;
    float s2 = c0 * a.z + c1 * bb.z + c2 * c.z + c3 * d.z;
    float s3 = c0 * a.w + c1 * bb.w + c2 * c.w + c3 * d.w;
    mx = fmaxf(mx, fmaxf(fmaxf(s0, s1), fmaxf(s2, s3)));
  }
  rm[idx] = fmaxf(mx, 1e-6f);
}

// ============ Kernel 3: flash attention with isolation bias ============
// block 256 (8 waves); one wave per (b, h, 16-query tile); loops 32-key chunks.
__global__ __launch_bounds__(256) void attn_kernel(
    const bf16* __restrict__ q_ws, const bf16* __restrict__ k_ws,
    const bf16* __restrict__ v_wsT, const float* __restrict__ cmask,
    const float* __restrict__ imask, const float* __restrict__ rm,
    const float* __restrict__ gate_in, bf16* __restrict__ ao)
{
  __shared__ __align__(64) bf16 lP[8][16][32];   // per-wave P transpose staging

  const int tid = threadIdx.x, wave = tid >> 5, lane = tid & 31;
  const int gw = blockIdx.x * 8 + wave;
  const int qt = gw % (N_ / 16);
  const int bh = gw / (N_ / 16);
  const int b = bh / H_, h = bh % H_;
  const int ar = lane & 15;
  const int kb0A = (lane < 16) ? 0 : 8;
  const int kbB  = (lane < 16) ? 0 : 8;
  const int hi8  = (lane >= 16) ? 8 : 0;

  float g = gate_in[h];
  const float gate3 = 3.f * fminf(fmaxf(g, 0.f), 1.f);

  // Q A-fragments for the two HD chunks (reused across all key chunks)
  const bf16* qrowp = q_ws + ((size_t)(bh * N_ + qt * 16 + ar)) * HD_;
  const v16bf qA0 = load_afrag(qrowp, kb0A);
  const v16bf qA1 = load_afrag(qrowp + 32, kb0A);

  float cmq[C_][8], invrm[8];
#pragma unroll
  for (int r = 0; r < 8; ++r) {
    const int qrow = qt * 16 + r + hi8;
    invrm[r] = 1.0f / rm[b * N_ + qrow];
#pragma unroll
    for (int c = 0; c < C_; ++c) cmq[c][r] = cmask[(b * C_ + c) * N_ + qrow];
  }

  float m[8], l[8];
  v8f oacc[4];
#pragma unroll
  for (int r = 0; r < 8; ++r) { m[r] = -1e30f; l[r] = 0.f; }
#pragma unroll
  for (int t = 0; t < 4; ++t) oacc[t] = vzero8();

  for (int kc = 0; kc < N_ / 32; ++kc) {
    const int k0 = kc * 32;
    // prefetch next chunk's K row and V rows into cache (global_prefetch_b8)
    if (kc + 1 < N_ / 32) {
      __builtin_prefetch(k_ws + ((size_t)(bh * N_ + k0 + 32 + ar)) * HD_, 0, 3);
      __builtin_prefetch(v_wsT + ((size_t)(bh * HD_ + ar)) * N_ + k0 + 32, 0, 3);
    }
    float S[2][8];
#pragma unroll
    for (int sub = 0; sub < 2; ++sub) {
      const int key = k0 + sub * 16 + ar;
      const bf16* krow = k_ws + ((size_t)(bh * N_ + key)) * HD_;
      v16bf kf0 = load_bfrag(krow, kbB);
      v16bf kf1 = load_bfrag(krow + 32, kbB);
      v8f s = vzero8();
      s = WMMA_BF16(qA0, kf0, s);
      s = WMMA_BF16(qA1, kf1, s);

      float cmk[C_];
#pragma unroll
      for (int c = 0; c < C_; ++c) cmk[c] = cmask[(b * C_ + c) * N_ + key];
#pragma unroll
      for (int r = 0; r < 8; ++r) {
        const int qrow = qt * 16 + r + hi8;
        float same = 0.f;
#pragma unroll
        for (int c = 0; c < C_; ++c) same += cmq[c][r] * cmk[c];
        float bias = (same * invrm[r] - 0.5f) * 2.0f +
                     BOOST_ * imask[((size_t)(b * N_ + qrow)) * N_ + key];
        S[sub][r] = s[r] * 0.125f + gate3 * bias;
      }
    }

    // online softmax across the 32 keys (rows live in 16-lane groups)
#pragma unroll
    for (int r = 0; r < 8; ++r) {
      float mx = fmaxf(S[0][r], S[1][r]);
      mx = fmaxf(mx, __shfl_xor(mx, 1, 32));
      mx = fmaxf(mx, __shfl_xor(mx, 2, 32));
      mx = fmaxf(mx, __shfl_xor(mx, 4, 32));
      mx = fmaxf(mx, __shfl_xor(mx, 8, 32));
      const float mnew = fmaxf(m[r], mx);
      const float fsc = __expf(m[r] - mnew);
      const float p0 = __expf(S[0][r] - mnew);
      const float p1 = __expf(S[1][r] - mnew);
      float ps = p0 + p1;
      ps += __shfl_xor(ps, 1, 32);
      ps += __shfl_xor(ps, 2, 32);
      ps += __shfl_xor(ps, 4, 32);
      ps += __shfl_xor(ps, 8, 32);
      l[r] = l[r] * fsc + ps;
      m[r] = mnew;
#pragma unroll
      for (int t = 0; t < 4; ++t) oacc[t][r] *= fsc;
      lP[wave][r + hi8][ar]      = (bf16)p0;    // D-layout -> LDS
      lP[wave][r + hi8][16 + ar] = (bf16)p1;
    }

    // per-wave DS ordering makes this read-after-write safe without a barrier
    const v16bf Pf = load_afrag(&lP[wave][ar][0], kb0A);
#pragma unroll
    for (int t = 0; t < 4; ++t) {
      const bf16* vrow = v_wsT + ((size_t)(bh * HD_ + t * 16 + ar)) * N_ + k0;
      v16bf vf = load_bfrag(vrow, kbB);
      oacc[t] = WMMA_BF16(Pf, vf, oacc[t]);
    }
  }

#pragma unroll
  for (int t = 0; t < 4; ++t)
#pragma unroll
    for (int r = 0; r < 8; ++r) {
      const int qrow = qt * 16 + r + hi8;
      ao[((size_t)(b * N_ + qrow)) * D_ + h * HD_ + t * 16 + ar] =
          (bf16)(oacc[t][r] / l[r]);
    }
}

// ============ Kernel 4: output projection (attn_out @ W_out^T) ============
// grid (B*N/16, D/128), block 256; wave w covers 16 output cols.
__global__ __launch_bounds__(256) void outproj_kernel(
    const bf16* __restrict__ ao, const float* __restrict__ Wout,
    float* __restrict__ out)
{
  const int tid = threadIdx.x, wave = tid >> 5, lane = tid & 31;
  const int ar = lane & 15;
  const int kb0A = (lane < 16) ? 0 : 8;
  const int kbB  = (lane < 16) ? 0 : 8;
  const int row  = blockIdx.x * 16 + ar;                   // A row for this lane
  const int col  = blockIdx.y * 128 + wave * 16 + ar;      // B col for this lane

  v8f acc = vzero8();
  const float* wrow = Wout + (size_t)col * D_;
  const bf16* arow = ao + (size_t)row * D_;
  for (int kk = 0; kk < D_ / 32; ++kk) {
    v16bf a = load_afrag(arow + kk * 32, kb0A);
    const float4* wp = (const float4*)(wrow + kk * 32 + kbB);
    float4 f0 = wp[0], f1 = wp[1], f2 = wp[4], f3 = wp[5];
    v16bf bfr;
    v4bf p0 = cvt4(f0), p1 = cvt4(f1), p2 = cvt4(f2), p3 = cvt4(f3);
#pragma unroll
    for (int e = 0; e < 4; ++e) {
      bfr[e] = p0[e]; bfr[4 + e] = p1[e]; bfr[8 + e] = p2[e]; bfr[12 + e] = p3[e];
    }
    acc = WMMA_BF16(a, bfr, acc);
  }
  const int hi8 = (lane >= 16) ? 8 : 0;
#pragma unroll
  for (int r = 0; r < 8; ++r)
    out[((size_t)(blockIdx.x * 16 + r + hi8)) * D_ +
        blockIdx.y * 128 + wave * 16 + ar] = acc[r];
}

extern "C" void kernel_launch(void* const* d_in, const int* in_sizes, int n_in,
                              void* d_out, int out_size, void* d_ws, size_t ws_size,
                              hipStream_t stream)
{
  (void)in_sizes; (void)n_in; (void)out_size; (void)ws_size;
  const float* x    = (const float*)d_in[0];
  const float* cm   = (const float*)d_in[1];
  const float* im   = (const float*)d_in[2];
  const float* Wqkv = (const float*)d_in[3];
  const float* Wout = (const float*)d_in[4];
  const float* qw   = (const float*)d_in[5];
  const float* kw   = (const float*)d_in[6];
  const float* gate = (const float*)d_in[7];
  float* out = (float*)d_out;

  char* ws = (char*)d_ws;
  bf16* q_ws  = (bf16*)(ws + OFF_Q);
  bf16* k_ws  = (bf16*)(ws + OFF_K);
  bf16* v_wsT = (bf16*)(ws + OFF_V);
  bf16* ao    = (bf16*)(ws + OFF_AO);
  float* rm   = (float*)(ws + OFF_RM);

  qkv_kernel<<<dim3(B_ * N_ / 16, H_), 128, 0, stream>>>(x, Wqkv, qw, kw,
                                                         q_ws, k_ws, v_wsT);
  rowmax_kernel<<<(B_ * N_) / 256, 256, 0, stream>>>(cm, rm);
  attn_kernel<<<(B_ * H_ * (N_ / 16)) / 8, 256, 0, stream>>>(q_ws, k_ws, v_wsT,
                                                             cm, im, rm, gate, ao);
  outproj_kernel<<<dim3((B_ * N_) / 16, D_ / 128), 256, 0, stream>>>(ao, Wout, out);
}
